// DecoupledPointJAFAR_52132313039153
// MI455X (gfx1250) — matrix operand order repaired and checked by reading, not compiled, not run on hardware
//
#include <hip/hip_runtime.h>
#include <hip/hip_bf16.h>

// ---------------------------------------------------------------------------
// DecoupledPointJAFAR forward for MI455X (gfx1250, wave32, WMMA).
//
// Heavy ops (64x64 channel GEMMs: ge2, q_w, k_w, rp2) are executed with
// v_wmma_f32_16x16x32_f16 (f16 inputs, f32 accumulate), K=64 = 2 chained
// WMMAs per 16x16 output tile, 4 N-tiles per 64-channel output.
// Small ops (ge1 6->64, FiLM, rp1 3->64, BN folds, bdy head, softmax(16),
// 13-way classifier) are VALU.
//
// Input order assumption: setup_inputs() dict insertion order, params
// flattened in insertion order (bn sub-dicts: gamma,beta,mean,var):
//  0 xyz  1 feat  2 k_idx
//  3 ge1_w 4 ge1_b 5..8 ge_bn1(g,b,m,v) 9 ge2_w 10 ge2_b 11..14 ge_bn2
//  15 scale_w 16 scale_b 17 shift_w 18 shift_b
//  19 q_w 20 q_b 21 k_w 22 k_b
//  23 bdy1_w 24 bdy1_b 25..28 bdy_bn 29 bdy2_w 30 bdy2_b
//  31 rp1_w 32 rp1_b 33..36 rp_bn 37 rp2_w 38 rp2_b 39 cls_w 40 cls_b
// Output order: logits(BN x 13) | affinity(BN x 16) | k_idx(BN x 16, float)
//               | out_feat(BN x 64) | bdy(BN)
// ---------------------------------------------------------------------------

#define BB   4
#define NN   8192
#define KNB  16
#define QK   64
#define CIN  6
#define NC   13
#define EPSB 1e-5f
#define NPTS (BB * NN)   // 32768

typedef _Float16 h16v __attribute__((ext_vector_type(16)));
typedef _Float16 h8v  __attribute__((ext_vector_type(8)));
typedef float    f8v  __attribute__((ext_vector_type(8)));

// A-matrix (16x32 f16) fragment: tile is [16][64] f16 row-major in LDS.
// lane<16: elems 0..15 = K kb..kb+15, kb=32*kc
// lane>=16: kb += 16.
__device__ __forceinline__ h16v ldsA(const _Float16* tile, int lane, int kc) {
  int m  = lane & 15;
  int kb = kc * 32 + ((lane >> 4) << 4);
  return *(const h16v*)(tile + m * QK + kb);
}

// B-matrix (32x16 f16) fragment for y = x @ W^T: B[k][n] = W[n][k].
// w is W (64x64, out-major) as f16 in LDS. lane column N = nt*16 + (lane&15),
// K = 32*kc + 16*(lane>=16) + e  -> 16 contiguous halves of row N.
__device__ __forceinline__ h16v ldsB(const _Float16* w, int lane, int kc, int nt) {
  int n  = nt * 16 + (lane & 15);
  int kb = kc * 32 + ((lane >> 4) << 4);
  return *(const h16v*)(w + n * QK + kb);
}

#define WMMA_F16(A, Bv, C) \
  __builtin_amdgcn_wmma_f32_16x16x32_f16(false, (A), false, (Bv), (short)0, (C), false, false)

// ===========================================================================
// Kernel 1: per-point pipeline. 256 thr = 8 waves, 16 points/wave,
// 128 points/block, grid = 256. Produces geom, Q, Kf (fp32 ws) + bdy logits.
// ===========================================================================
__global__ __launch_bounds__(256) void k_point(
    const float* __restrict__ feat,
    const float* __restrict__ ge1_w, const float* __restrict__ ge1_b,
    const float* __restrict__ bn1g, const float* __restrict__ bn1b,
    const float* __restrict__ bn1m, const float* __restrict__ bn1v,
    const float* __restrict__ ge2_w, const float* __restrict__ ge2_b,
    const float* __restrict__ bn2g, const float* __restrict__ bn2b,
    const float* __restrict__ bn2m, const float* __restrict__ bn2v,
    const float* __restrict__ scale_w, const float* __restrict__ scale_b,
    const float* __restrict__ shift_w, const float* __restrict__ shift_b,
    const float* __restrict__ q_w, const float* __restrict__ q_b,
    const float* __restrict__ k_w, const float* __restrict__ k_b,
    const float* __restrict__ bdy1_w, const float* __restrict__ bdy1_b,
    const float* __restrict__ bnbg, const float* __restrict__ bnbb,
    const float* __restrict__ bnbm, const float* __restrict__ bnbv,
    const float* __restrict__ bdy2_w, const float* __restrict__ bdy2_b,
    float* __restrict__ geomBuf, float* __restrict__ QBuf,
    float* __restrict__ KBuf, float* __restrict__ bdyOut)
{
  __shared__ _Float16 sWge2[QK * QK];
  __shared__ _Float16 sWq[QK * QK];
  __shared__ _Float16 sWk[QK * QK];
  __shared__ _Float16 sTile[8][16 * QK];
  __shared__ float sFeat[8][16 * CIN];
  __shared__ float sGe1w[QK * CIN], sSclw[QK * CIN], sShfw[QK * CIN];
  __shared__ float sBdy1w[32 * QK], sBdy2w[32];
  __shared__ float sA1[QK], sC1[QK], sA2[QK], sC2[QK], sAb[32], sCb[32];
  __shared__ float sGe1b[QK], sSclb[QK], sShfb[QK], sBdy1b[32];

  const int tid = threadIdx.x;
  for (int i = tid; i < QK * QK; i += 256) {
    sWge2[i] = (_Float16)ge2_w[i];
    sWq[i]   = (_Float16)q_w[i];
    sWk[i]   = (_Float16)k_w[i];
  }
  for (int i = tid; i < QK * CIN; i += 256) {
    sGe1w[i] = ge1_w[i]; sSclw[i] = scale_w[i]; sShfw[i] = shift_w[i];
  }
  for (int i = tid; i < 32 * QK; i += 256) sBdy1w[i] = bdy1_w[i];
  if (tid < 32) {
    float a = bnbg[tid] * rsqrtf(bnbv[tid] + EPSB);
    sAb[tid] = a; sCb[tid] = bnbb[tid] - bnbm[tid] * a;
    sBdy2w[tid] = bdy2_w[tid]; sBdy1b[tid] = bdy1_b[tid];
  }
  if (tid < QK) {
    float a1 = bn1g[tid] * rsqrtf(bn1v[tid] + EPSB);
    sA1[tid] = a1; sC1[tid] = bn1b[tid] - bn1m[tid] * a1;
    float a2 = bn2g[tid] * rsqrtf(bn2v[tid] + EPSB);
    sA2[tid] = a2; sC2[tid] = bn2b[tid] - bn2m[tid] * a2;
    sGe1b[tid] = ge1_b[tid]; sSclb[tid] = scale_b[tid]; sShfb[tid] = shift_b[tid];
  }
  __syncthreads();

  const int wave = tid >> 5, lane = tid & 31;
  const int ptb  = blockIdx.x * 128 + wave * 16;   // global point base of tile
  _Float16* tile = sTile[wave];
  float* featT   = sFeat[wave];

  // stage feat tile (16 x 6)
  for (int i = lane; i < 16 * CIN; i += 32)
    featT[i] = feat[(size_t)(ptb + i / CIN) * CIN + (i % CIN)];
  __syncthreads();

  // ge1 + bn1 + relu -> f16 tile (16 pts x 64 ch)
#pragma unroll
  for (int it = 0; it < 32; ++it) {
    int v = it * 32 + lane; int p = v >> 6; int c = v & 63;
    float x = sGe1b[c];
#pragma unroll
    for (int j = 0; j < CIN; ++j) x += featT[p * CIN + j] * sGe1w[c * CIN + j];
    x = fmaxf(x * sA1[c] + sC1[c], 0.f);
    tile[p * QK + c] = (_Float16)x;
  }
  __syncthreads();

  // ge2 (WMMA) + bn2 + relu + FiLM
  h16v a0 = ldsA(tile, lane, 0);
  h16v a1 = ldsA(tile, lane, 1);
  const int col = lane & 15, hibase = (lane >> 4) << 3;
  float geomReg[4][8];
#pragma unroll
  for (int t = 0; t < 4; ++t) {
    int n = t * 16 + col;
    f8v c; float bias = ge2_b[n];
#pragma unroll
    for (int r = 0; r < 8; ++r) c[r] = bias;
    c = WMMA_F16(a0, ldsB(sWge2, lane, 0, t), c);
    c = WMMA_F16(a1, ldsB(sWge2, lane, 1, t), c);
    float sclb = sSclb[n], shfb = sShfb[n];
#pragma unroll
    for (int r = 0; r < 8; ++r) {
      int m = r + hibase;
      float s = sclb, sh = shfb;
#pragma unroll
      for (int j = 0; j < CIN; ++j) {
        float fj = featT[m * CIN + j];
        s  += fj * sSclw[n * CIN + j];
        sh += fj * sShfw[n * CIN + j];
      }
      float g0 = fmaxf(c[r] * sA2[n] + sC2[n], 0.f);
      geomReg[t][r] = g0 * (s + 1.f) + sh;
    }
  }
  __syncthreads();

  // write geom: f16 tile (reused) + fp32 global
#pragma unroll
  for (int t = 0; t < 4; ++t) {
    int n = t * 16 + col;
#pragma unroll
    for (int r = 0; r < 8; ++r) {
      int m = r + hibase;
      float g = geomReg[t][r];
      tile[m * QK + n] = (_Float16)g;
      geomBuf[(size_t)(ptb + m) * QK + n] = g;
    }
  }
  __syncthreads();

  // boundary head: lanes split the 32 hidden units, pair-reduce via shfl
  {
    int p = col, half = lane >> 4;
    float ssum = 0.f;
    for (int o = half * 16; o < half * 16 + 16; ++o) {
      float acc = sBdy1b[o];
      for (int cch = 0; cch < QK; ++cch)
        acc += (float)tile[p * QK + cch] * sBdy1w[o * QK + cch];
      float h2 = fmaxf(acc * sAb[o] + sCb[o], 0.f);
      ssum += h2 * sBdy2w[o];
    }
    float other = __shfl_xor(ssum, 16, 32);
    if (half == 0) bdyOut[ptb + p] = ssum + other + bdy2_b[0];
  }

  // Q / Kf projections (WMMA)
  h16v g0 = ldsA(tile, lane, 0);
  h16v g1 = ldsA(tile, lane, 1);
#pragma unroll
  for (int t = 0; t < 4; ++t) {
    int n = t * 16 + col;
    f8v cq, ck; float qb = q_b[n], kb = k_b[n];
#pragma unroll
    for (int r = 0; r < 8; ++r) { cq[r] = qb; ck[r] = kb; }
    cq = WMMA_F16(g0, ldsB(sWq, lane, 0, t), cq);
    cq = WMMA_F16(g1, ldsB(sWq, lane, 1, t), cq);
    ck = WMMA_F16(g0, ldsB(sWk, lane, 0, t), ck);
    ck = WMMA_F16(g1, ldsB(sWk, lane, 1, t), ck);
#pragma unroll
    for (int r = 0; r < 8; ++r) {
      int m = r + hibase;
      QBuf[(size_t)(ptb + m) * QK + n] = cq[r];
      KBuf[(size_t)(ptb + m) * QK + n] = ck[r];
    }
  }
}

// ===========================================================================
// Kernel 2: per-point attention. 1 wave = 1 point (its 16 neighbors form one
// 16x64 WMMA A-tile for rp2). 8 points/block, grid = 4096.
// ===========================================================================
__global__ __launch_bounds__(256) void k_attn(
    const float* __restrict__ xyz, const int* __restrict__ kidx,
    const float* __restrict__ rp1_w, const float* __restrict__ rp1_b,
    const float* __restrict__ bng, const float* __restrict__ bnb,
    const float* __restrict__ bnm, const float* __restrict__ bnv,
    const float* __restrict__ rp2_w, const float* __restrict__ rp2_b,
    const float* __restrict__ cls_w, const float* __restrict__ cls_b,
    const float* __restrict__ geomBuf, const float* __restrict__ QBuf,
    const float* __restrict__ KBuf,
    float* __restrict__ outLogits, float* __restrict__ outAff,
    float* __restrict__ outIdx, float* __restrict__ outFeat)
{
  __shared__ _Float16 sWrp2[QK * QK];
  __shared__ float sRp1w[QK * 3], sRp1b[QK], sA[QK], sC[QK];
  __shared__ float sClsw[NC * QK], sClsb[NC];
  __shared__ int   sIdx[8][16];
  __shared__ float sRel[8][16][3];
  __shared__ _Float16 sTP[8][16 * QK];
  __shared__ float sRed[8][16][16];
  __shared__ float sLg[8][16];
  __shared__ float sOf[8][QK][2];
  __shared__ float sOfF[8][QK];

  const int tid = threadIdx.x;
  for (int i = tid; i < QK * QK; i += 256) sWrp2[i] = (_Float16)rp2_w[i];
  for (int i = tid; i < QK * 3; i += 256)  sRp1w[i] = rp1_w[i];
  for (int i = tid; i < NC * QK; i += 256) sClsw[i] = cls_w[i];
  if (tid < QK) {
    float a = bng[tid] * rsqrtf(bnv[tid] + EPSB);
    sA[tid] = a; sC[tid] = bnb[tid] - bnm[tid] * a;
    sRp1b[tid] = rp1_b[tid];
  }
  if (tid < NC) sClsb[tid] = cls_b[tid];
  __syncthreads();

  const int wave = tid >> 5, lane = tid & 31;
  const int pt  = blockIdx.x * 8 + wave;   // 0..B*N-1
  const int bb0 = (pt >> 13) << 13;        // batch base point index

  if (lane < 16) {
    int idx = kidx[(size_t)pt * KNB + lane];
    sIdx[wave][lane] = idx;
    outIdx[(size_t)pt * KNB + lane] = (float)idx;
#pragma unroll
    for (int j = 0; j < 3; ++j)
      sRel[wave][lane][j] = xyz[(size_t)pt * 3 + j] - xyz[(size_t)(bb0 + idx) * 3 + j];
  }
  __syncthreads();

  // rp1 + bn + relu -> f16 tile (16 neighbors x 64 ch)
  _Float16* tP = sTP[wave];
#pragma unroll
  for (int it = 0; it < 32; ++it) {
    int v = it * 32 + lane; int m = v >> 6; int c = v & 63;
    float x = sRp1b[c];
#pragma unroll
    for (int j = 0; j < 3; ++j) x += sRel[wave][m][j] * sRp1w[c * 3 + j];
    x = fmaxf(x * sA[c] + sC[c], 0.f);
    tP[m * QK + c] = (_Float16)x;
  }
  __syncthreads();

  // rp2 (WMMA): pos_enc lives in acc, layout m = r + 8*hi, c = 16t + col
  h16v a0 = ldsA(tP, lane, 0);
  h16v a1 = ldsA(tP, lane, 1);
  const int col = lane & 15, hi = lane >> 4;
  f8v acc[4];
#pragma unroll
  for (int t = 0; t < 4; ++t) {
    int n = t * 16 + col;
    float bias = rp2_b[n];
#pragma unroll
    for (int r = 0; r < 8; ++r) acc[t][r] = bias;
    acc[t] = WMMA_F16(a0, ldsB(sWrp2, lane, 0, t), acc[t]);
    acc[t] = WMMA_F16(a1, ldsB(sWrp2, lane, 1, t), acc[t]);
  }

  // attn partials: Q . (Kf_gathered + pos_enc); reduce 16 lanes per neighbor
  float qv[4];
#pragma unroll
  for (int t = 0; t < 4; ++t) qv[t] = QBuf[(size_t)pt * QK + t * 16 + col];
#pragma unroll
  for (int r = 0; r < 8; ++r) {
    int m = r + (hi << 3);
    size_t gb = (size_t)(bb0 + sIdx[wave][m]) * QK;
    float part = 0.f;
#pragma unroll
    for (int t = 0; t < 4; ++t)
      part += qv[t] * (KBuf[gb + t * 16 + col] + acc[t][r]);
    sRed[wave][m][col] = part;
  }
  __syncthreads();

  if (lane < 16) {
    float s = 0.f;
    for (int i = 0; i < 16; ++i) s += sRed[wave][lane][i];
    sLg[wave][lane] = s * 0.125f;   // / sqrt(64)
  }
  __syncthreads();

  // softmax over 16 (redundant per lane; trivial)
  float mx = -1e30f;
  for (int i = 0; i < 16; ++i) mx = fmaxf(mx, sLg[wave][i]);
  float den = 0.f;
  for (int i = 0; i < 16; ++i) den += __expf(sLg[wave][i] - mx);
  float inv = 1.f / den;
  if (lane < 16) outAff[(size_t)pt * KNB + lane] = __expf(sLg[wave][lane] - mx) * inv;

  // out_feat = sum_m aff[m] * geom[idx[m]]
  float ofp[4] = {0.f, 0.f, 0.f, 0.f};
#pragma unroll
  for (int r = 0; r < 8; ++r) {
    int m = r + (hi << 3);
    float aw = __expf(sLg[wave][m] - mx) * inv;
    size_t gb = (size_t)(bb0 + sIdx[wave][m]) * QK;
#pragma unroll
    for (int t = 0; t < 4; ++t) ofp[t] += aw * geomBuf[gb + t * 16 + col];
  }
#pragma unroll
  for (int t = 0; t < 4; ++t) sOf[wave][t * 16 + col][hi] = ofp[t];
  __syncthreads();

#pragma unroll
  for (int cc = 0; cc < 2; ++cc) {
    int c = lane + cc * 32;
    float v = sOf[wave][c][0] + sOf[wave][c][1];
    sOfF[wave][c] = v;
    outFeat[(size_t)pt * QK + c] = v;
  }
  __syncthreads();

  // classifier: 13 outputs, one lane each
  if (lane < NC) {
    float s = sClsb[lane];
    for (int c = 0; c < QK; ++c) s += sOfF[wave][c] * sClsw[lane * QK + c];
    outLogits[(size_t)pt * NC + lane] = s;
  }
}

// ===========================================================================
extern "C" void kernel_launch(void* const* d_in, const int* in_sizes, int n_in,
                              void* d_out, int out_size, void* d_ws, size_t ws_size,
                              hipStream_t stream) {
  const float* xyz     = (const float*)d_in[0];
  const float* feat    = (const float*)d_in[1];
  const int*   kidx    = (const int*)d_in[2];
  const float* ge1_w   = (const float*)d_in[3];
  const float* ge1_b   = (const float*)d_in[4];
  const float* bn1g    = (const float*)d_in[5];
  const float* bn1b    = (const float*)d_in[6];
  const float* bn1m    = (const float*)d_in[7];
  const float* bn1v    = (const float*)d_in[8];
  const float* ge2_w   = (const float*)d_in[9];
  const float* ge2_b   = (const float*)d_in[10];
  const float* bn2g    = (const float*)d_in[11];
  const float* bn2b    = (const float*)d_in[12];
  const float* bn2m    = (const float*)d_in[13];
  const float* bn2v    = (const float*)d_in[14];
  const float* scale_w = (const float*)d_in[15];
  const float* scale_b = (const float*)d_in[16];
  const float* shift_w = (const float*)d_in[17];
  const float* shift_b = (const float*)d_in[18];
  const float* q_w     = (const float*)d_in[19];
  const float* q_b     = (const float*)d_in[20];
  const float* k_w     = (const float*)d_in[21];
  const float* k_b     = (const float*)d_in[22];
  const float* bdy1_w  = (const float*)d_in[23];
  const float* bdy1_b  = (const float*)d_in[24];
  const float* bnbg    = (const float*)d_in[25];
  const float* bnbb    = (const float*)d_in[26];
  const float* bnbm    = (const float*)d_in[27];
  const float* bnbv    = (const float*)d_in[28];
  const float* bdy2_w  = (const float*)d_in[29];
  const float* bdy2_b  = (const float*)d_in[30];
  const float* rp1_w   = (const float*)d_in[31];
  const float* rp1_b   = (const float*)d_in[32];
  const float* bnrg    = (const float*)d_in[33];
  const float* bnrb    = (const float*)d_in[34];
  const float* bnrm    = (const float*)d_in[35];
  const float* bnrv    = (const float*)d_in[36];
  const float* rp2_w   = (const float*)d_in[37];
  const float* rp2_b   = (const float*)d_in[38];
  const float* cls_w   = (const float*)d_in[39];
  const float* cls_b   = (const float*)d_in[40];

  // workspace: geom | Q | Kf  (fp32, 8 MB each)
  float* ws      = (float*)d_ws;
  float* geomBuf = ws;
  float* QBuf    = geomBuf + (size_t)NPTS * QK;
  float* KBuf    = QBuf    + (size_t)NPTS * QK;

  // outputs: logits | affinity | k_idx | out_feat | bdy
  float* out       = (float*)d_out;
  float* outLogits = out;
  float* outAff    = outLogits + (size_t)NPTS * NC;
  float* outIdx    = outAff    + (size_t)NPTS * KNB;
  float* outFeat   = outIdx    + (size_t)NPTS * KNB;
  float* outBdy    = outFeat   + (size_t)NPTS * QK;

  k_point<<<NPTS / 128, 256, 0, stream>>>(
      feat, ge1_w, ge1_b, bn1g, bn1b, bn1m, bn1v,
      ge2_w, ge2_b, bn2g, bn2b, bn2m, bn2v,
      scale_w, scale_b, shift_w, shift_b,
      q_w, q_b, k_w, k_b,
      bdy1_w, bdy1_b, bnbg, bnbb, bnbm, bnbv, bdy2_w, bdy2_b,
      geomBuf, QBuf, KBuf, outBdy);

  k_attn<<<NPTS / 8, 256, 0, stream>>>(
      xyz, kidx, rp1_w, rp1_b, bnrg, bnrb, bnrm, bnrv,
      rp2_w, rp2_b, cls_w, cls_b,
      geomBuf, QBuf, KBuf,
      outLogits, outAff, outIdx, outFeat);
}
